// Full_U_Mamba_Block_69776038691244
// MI455X (gfx1250) — matrix operand order repaired and checked by reading, not compiled, not run on hardware
//
#include <hip/hip_runtime.h>
#include <hip/hip_bf16.h>

typedef __attribute__((ext_vector_type(16))) _Float16 v16h;
typedef __attribute__((ext_vector_type(8)))  float    v8f;
typedef _Float16 h16;

static constexpr int M = 65536;   // B*H*W*D tokens (2*32*32*32)

#define DEV __device__ __forceinline__
DEV float sigmoidf_(float x){ return 1.f/(1.f+__expf(-x)); }
DEV float siluf_(float x){ return x*sigmoidf_(x); }
DEV float softplusf_(float x){ return (x > 20.f) ? x : __logf(1.f + __expf(x)); }

// ---------------- layout: NCDHW f32 -> channel-last f32 + f16 ----------------
__global__ void k_to_cl(const float* __restrict__ x, float* __restrict__ xcl, h16* __restrict__ xh){
  int i = blockIdx.x*256 + threadIdx.x;            // over M*64, channel-last index
  if (i >= M*64) return;
  int c = i & 63; int t = i >> 6;
  int b = t >> 15; int sp = t & 32767;
  float v = x[((size_t)(b*64 + c) << 15) + sp];
  xcl[i] = v; xh[i] = (h16)v;
}

// ---------------- weight prep: convert/transpose all GEMM weights to f16 ----------------
__global__ void k_prep(const float* __restrict__ w1, const float* __restrict__ w2,
                       const float* __restrict__ lw, const float* __restrict__ rw,
                       const float* __restrict__ dw, const float* __restrict__ bw,
                       const float* __restrict__ cw, const float* __restrict__ ow,
                       h16* __restrict__ wtap, h16* __restrict__ lrw,
                       h16* __restrict__ wd, h16* __restrict__ outw){
  int i = blockIdx.x*256 + threadIdx.x;
  if (i < 2*27*64*64){                                  // wtap[stage][tap][co][ci]
    int s = i / (27*64*64); int r = i % (27*64*64);
    int tap = r >> 12; int co = (r & 4095) >> 6; int ci = r & 63;
    const float* src = s ? w2 : w1;
    wtap[((size_t)s*27 + tap)*4096 + co*64 + ci] = (h16)src[(co*64 + ci)*27 + tap];
  }
  int j = i - 2*27*64*64;
  if (j >= 0 && j < 256*64){                            // [left_w ; right_w]  (N=256,K=64)
    int n = j >> 6, k = j & 63;
    lrw[j] = (h16)((n < 128) ? lw[n*64 + k] : rw[(n-128)*64 + k]);
  }
  int q = j - 256*64;
  if (q >= 0 && q < 160*128){                           // [delta_w ; Bp_w ; Cp_w] (N=160,K=128)
    int n = q >> 7, k = q & 127;
    float v = (n < 128) ? dw[n*128 + k] : (n < 144 ? bw[(n-128)*128 + k] : cw[(n-144)*128 + k]);
    wd[q] = (h16)v;
  }
  int u = q - 160*128;
  if (u >= 0 && u < 64*128) outw[u] = (h16)ow[u];       // out_w (N=64,K=128)
}

// ---------------- Conv3d k=3 as implicit GEMM with WMMA ----------------
// one block per (b,h,w); LDS slab [3h][3w][34d][64c] f16; 8 waves = 2 d-tiles x 4 co-tiles
__global__ void __launch_bounds__(256) k_conv(const h16* __restrict__ xh, const h16* __restrict__ wt,
                                              const float* __restrict__ bias, float* __restrict__ raw){
  __shared__ __align__(16) h16 slab[3*3*34*64];
  int blk = blockIdx.x;                      // b*1024 + h*32 + w
  int b = blk >> 10; int h = (blk >> 5) & 31; int w = blk & 31;
  int tokenbase = blk * 32;
  const unsigned int* src32 = (const unsigned int*)xh;
  unsigned int* dst32 = (unsigned int*)slab;
  for (int idx = threadIdx.x; idx < 3*3*34*32; idx += 256){
    int c2 = idx & 31;
    int t3 = idx >> 5;
    int dz = t3 % 34; int t4 = t3 / 34; int ww = t4 % 3; int hh = t4 / 3;
    int hi = h + hh - 1, wi = w + ww - 1, dd = dz - 1;
    unsigned int val = 0u;
    if ((unsigned)hi < 32u && (unsigned)wi < 32u && (unsigned)dd < 32u){
      int tok = (b << 15) + (hi << 10) + (wi << 5) + dd;
      val = src32[(size_t)tok*32 + c2];
    }
    dst32[idx] = val;
  }
  __syncthreads();
  int lane = threadIdx.x & 31;
  int wv   = threadIdx.x >> 5;
  int mtile  = wv & 1;                 // d in [mtile*16, +16)
  int cobase = (wv >> 1) * 16;         // co tile
  int r   = lane & 15;
  int kbA = (lane & 16) ? 8 : 0;       // A frag: runs at k = kbA.. and kbA+16..
  int kbB = (lane & 16) ? 8 : 0;       // B frag: runs at k = kbB.. and kbB+16..
  v8f acc = {};
  for (int kh = 0; kh < 3; kh++)
    for (int kw = 0; kw < 3; kw++)
      for (int kd = 0; kd < 3; kd++){
        int tap = (kh*3 + kw)*3 + kd;
        int dpos = mtile*16 + r + kd;                 // halo-adjusted d index [0,33]
        int rowbase = ((kh*3 + kw)*34 + dpos) * 64;
#pragma unroll
        for (int kc = 0; kc < 2; kc++){
          const uint4* pa = reinterpret_cast<const uint4*>(slab) + ((rowbase + kc*32 + kbA) >> 3);
          v16h a; reinterpret_cast<uint4*>(&a)[0] = pa[0]; reinterpret_cast<uint4*>(&a)[1] = pa[2];
          const uint4* pb = reinterpret_cast<const uint4*>(
              wt + ((size_t)tap*64 + cobase + (lane & 15))*64 + kc*32 + kbB);
          v16h bq; reinterpret_cast<uint4*>(&bq)[0] = pb[0]; reinterpret_cast<uint4*>(&bq)[1] = pb[2];
          acc = __builtin_amdgcn_wmma_f32_16x16x32_f16(false, a, false, bq, (short)0, acc, false, false);
        }
      }
  int co = cobase + (lane & 15);
  float bv = bias[co];
  int mofs = (lane & 16) ? 8 : 0;
#pragma unroll
  for (int r8 = 0; r8 < 8; r8++){
    int d = mtile*16 + r8 + mofs;
    raw[(size_t)(tokenbase + d)*64 + co] = acc[r8] + bv;
  }
}

// ---------------- InstanceNorm stats: per (b,c) mean & rstd over 32768 voxels ----------------
__global__ void k_stats(const float* __restrict__ raw, float* __restrict__ stats){
  __shared__ float s1[256], s2[256];
  int bc = blockIdx.x;                 // b*64 + c
  int b = bc >> 6, c = bc & 63;
  const float* p = raw + ((size_t)b << 15)*64 + c;
  float sum = 0.f, sq = 0.f;
  for (int sp = threadIdx.x; sp < 32768; sp += 256){
    float v = p[(size_t)sp*64];
    sum += v; sq += v*v;
  }
  s1[threadIdx.x] = sum; s2[threadIdx.x] = sq; __syncthreads();
  for (int o = 128; o > 0; o >>= 1){
    if (threadIdx.x < o){ s1[threadIdx.x] += s1[threadIdx.x+o]; s2[threadIdx.x] += s2[threadIdx.x+o]; }
    __syncthreads();
  }
  if (threadIdx.x == 0){
    float mu = s1[0] * (1.f/32768.f);
    float var = s2[0] * (1.f/32768.f) - mu*mu;
    stats[bc*2] = mu; stats[bc*2+1] = rsqrtf(var + 1e-5f);
  }
}

// norm + LeakyReLU + residual (in place), refresh f16 shadow
__global__ void k_conv_fin(const float* __restrict__ raw, const float* __restrict__ stats,
                           float* __restrict__ xcl, h16* __restrict__ xh){
  int i = blockIdx.x*256 + threadIdx.x;
  if (i >= M*64) return;
  int c = i & 63; int t = i >> 6; int b = t >> 15;
  float mu = stats[(b*64+c)*2], rs = stats[(b*64+c)*2+1];
  float y = (raw[i] - mu) * rs;
  y = (y >= 0.f) ? y : 0.01f * y;
  float v = xcl[i] + y;
  xcl[i] = v; xh[i] = (h16)v;
}

// ---------------- LayerNorm over C=64 (shared by all three axes) ----------------
__global__ void k_ln64(const float* __restrict__ xcl, const float* __restrict__ lnw,
                       const float* __restrict__ lnb, h16* __restrict__ outh){
  int t = blockIdx.x*256 + threadIdx.x;
  if (t >= M) return;
  const float* p = xcl + (size_t)t*64;
  float sum = 0.f, sq = 0.f;
  for (int k = 0; k < 64; k++){ float v = p[k]; sum += v; sq += v*v; }
  float mu = sum * (1.f/64.f);
  float rs = rsqrtf(sq * (1.f/64.f) - mu*mu + 1e-5f);
  for (int k = 0; k < 64; k++)
    outh[(size_t)t*64 + k] = (h16)(((p[k]-mu)*rs)*lnw[k] + lnb[k]);
}

// ---------------- generic WMMA GEMM:  C[M,N] = A[M,K](f16) * W[N,K]^T(f16) ----------------
template<int K, int N>
__global__ void __launch_bounds__(256) k_gemm(const h16* __restrict__ A, const h16* __restrict__ W,
                                              float* __restrict__ C){
  __shared__ __align__(16) h16 wl[N*K];
  for (int i = threadIdx.x; i < (N*K)/8; i += 256)
    reinterpret_cast<uint4*>(wl)[i] = reinterpret_cast<const uint4*>(W)[i];
  __syncthreads();
  int lane = threadIdx.x & 31, wv = threadIdx.x >> 5;
  int mbase = blockIdx.x*128 + wv*16;
  int r   = lane & 15;
  int kbA = (lane & 16) ? 8 : 0;
  int kbB = (lane & 16) ? 8 : 0;
  v8f acc[N/16] = {};
#pragma unroll
  for (int kc = 0; kc < K/32; kc++){
    const uint4* pa = reinterpret_cast<const uint4*>(A + (size_t)(mbase + r)*K + kc*32 + kbA);
    v16h a; reinterpret_cast<uint4*>(&a)[0] = pa[0]; reinterpret_cast<uint4*>(&a)[1] = pa[2];
#pragma unroll
    for (int j = 0; j < N/16; j++){
      const uint4* pb = reinterpret_cast<const uint4*>(wl + (j*16 + (lane & 15))*K + kc*32 + kbB);
      v16h bq; reinterpret_cast<uint4*>(&bq)[0] = pb[0]; reinterpret_cast<uint4*>(&bq)[1] = pb[2];
      acc[j] = __builtin_amdgcn_wmma_f32_16x16x32_f16(false, a, false, bq, (short)0, acc[j], false, false);
    }
  }
  int mofs = (lane & 16) ? 8 : 0;
#pragma unroll
  for (int j = 0; j < N/16; j++)
#pragma unroll
    for (int r8 = 0; r8 < 8; r8++)
      C[(size_t)(mbase + r8 + mofs)*N + j*16 + (lane & 15)] = acc[j][r8];
}

// silu the gate half of LR in place (cols 128..255)
__global__ void k_silu_gate(float* __restrict__ lr){
  size_t i = (size_t)blockIdx.x*256 + threadIdx.x;     // over M*128
  if (i >= (size_t)M*128) return;
  size_t t = i >> 7; int k = (int)(i & 127);
  float* p = lr + t*256 + 128 + k;
  *p = siluf_(*p);
}

// causal depthwise conv1d (dc=4) along axis + bias + silu  -> f16
__global__ void k_conv1d(const float* __restrict__ lr, const float* __restrict__ cw,
                         const float* __restrict__ cb, h16* __restrict__ xlh, int stride){
  int i = blockIdx.x*256 + threadIdx.x;                // over M*128
  if (i >= M*128) return;
  int ch = i & 127; int t = i >> 7;
  int l = (t / stride) & 31;
  float acc = cb[ch];
#pragma unroll
  for (int j = 0; j < 4; j++){
    int ll = l + j - 3;
    if (ll >= 0) acc += lr[(size_t)(t + (j-3)*stride)*256 + ch] * cw[ch*4 + j];
  }
  xlh[(size_t)t*128 + ch] = (h16)siluf_(acc);
}

// selective scan: block per sequence (n), thread per d_inner channel, 16-state recurrence
__global__ void __launch_bounds__(128) k_scan(const float* __restrict__ dbc, const h16* __restrict__ xlh,
                                              const float* __restrict__ delta_b, const float* __restrict__ A_log,
                                              float* __restrict__ Y, int axis){
  __shared__ float bcs[32];
  int n  = blockIdx.x;
  int di = threadIdx.x;
  int base, stride;
  if (axis == 0){ base = ((n >> 10) << 15) + (n & 1023); stride = 1024; }   // scan over H
  else if (axis == 1){ base = ((n >> 5) << 10) + (n & 31); stride = 32; }   // scan over W
  else { base = n << 5; stride = 1; }                                       // scan over D
  float As[16];
#pragma unroll
  for (int s = 0; s < 16; s++) As[s] = -softplusf_(A_log[s]);
  float db = delta_b[di];
  float hst[16];
#pragma unroll
  for (int s = 0; s < 16; s++) hst[s] = 0.f;
  for (int l = 0; l < 32; l++){
    int t = base + l*stride;
    if (di < 32) bcs[di] = dbc[(size_t)t*160 + 128 + di];     // B (0..15), C (16..31)
    __syncthreads();
    float delta = softplusf_(dbc[(size_t)t*160 + di] + db);
    delta = fminf(fmaxf(delta, 1e-4f), 10.f);
    float xv = (float)xlh[(size_t)t*128 + di];
    float du = delta * xv;
    float y = 0.f;
#pragma unroll
    for (int s = 0; s < 16; s++){
      float ab = __expf(delta * As[s]);
      hst[s] = ab*hst[s] + du*bcs[s];
      y += hst[s] * bcs[16+s];
    }
    Y[(size_t)t*128 + di] = y;
    __syncthreads();
  }
}

// z = LayerNorm_128(y * gate) with pn_w/pn_b  -> f16
__global__ void k_pn(const float* __restrict__ Y, const float* __restrict__ lr,
                     const float* __restrict__ pw, const float* __restrict__ pb, h16* __restrict__ zh){
  int t = blockIdx.x*256 + threadIdx.x;
  if (t >= M) return;
  const float* yp = Y  + (size_t)t*128;
  const float* gp = lr + (size_t)t*256 + 128;
  float sum = 0.f, sq = 0.f;
  for (int k = 0; k < 128; k++){ float v = yp[k]*gp[k]; sum += v; sq += v*v; }
  float mu = sum * (1.f/128.f);
  float rs = rsqrtf(sq * (1.f/128.f) - mu*mu + 1e-5f);
  for (int k = 0; k < 128; k++){
    float v = yp[k]*gp[k];
    zh[(size_t)t*128 + k] = (h16)((v-mu)*rs*pw[k] + pb[k]);
  }
}

// acc (+)= softmax(axis_w)[axis] * m
__global__ void k_axpy(const float* __restrict__ m, const float* __restrict__ axw,
                       float* __restrict__ acc, int axis, int first){
  int i = blockIdx.x*256 + threadIdx.x;
  if (i >= M*64) return;
  float e0 = __expf(axw[0]), e1 = __expf(axw[1]), e2 = __expf(axw[2]);
  float wa = ((axis == 0) ? e0 : (axis == 1) ? e1 : e2) / (e0 + e1 + e2);
  float v = wa * m[i];
  acc[i] = first ? v : (acc[i] + v);
}

// out(NCDHW) = identity + res_scale * acc(channel-last)
__global__ void k_final(const float* __restrict__ x, const float* __restrict__ acc,
                        const float* __restrict__ rs, float* __restrict__ out){
  int i = blockIdx.x*256 + threadIdx.x;
  if (i >= M*64) return;
  int sp = i & 32767; int bc = i >> 15; int c = bc & 63; int b = bc >> 6;
  int t = (b << 15) + sp;
  out[i] = x[i] + rs[0] * acc[(size_t)t*64 + c];
}

extern "C" void kernel_launch(void* const* d_in, const int* in_sizes, int n_in,
                              void* d_out, int out_size, void* d_ws, size_t ws_size,
                              hipStream_t stream){
  (void)in_sizes; (void)n_in; (void)out_size;
  const float* x      = (const float*)d_in[0];
  const float* cr1_w  = (const float*)d_in[1];
  const float* cr1_b  = (const float*)d_in[2];
  const float* cr2_w  = (const float*)d_in[3];
  const float* cr2_b  = (const float*)d_in[4];
  const float* ln_w   = (const float*)d_in[5];
  const float* ln_b   = (const float*)d_in[6];
  // 7: left_w
  const float* c1w    = (const float*)d_in[8];
  const float* c1b    = (const float*)d_in[9];
  // 10: delta_w
  const float* delta_b= (const float*)d_in[11];
  // 12: Bp_w, 13: Cp_w
  const float* A_log  = (const float*)d_in[14];
  // 15: right_w
  const float* pn_w   = (const float*)d_in[16];
  const float* pn_b   = (const float*)d_in[17];
  // 18: out_w
  const float* res_s  = (const float*)d_in[19];
  const float* axis_w = (const float*)d_in[20];
  float* out = (float*)d_out;

  char* base = (char*)d_ws;
  size_t off = 0;
  auto alloc = [&](size_t bytes)->char*{ char* r = base + off; off += (bytes + 255) & ~(size_t)255; return r; };
  float* XCL  = (float*)alloc((size_t)M*64*4);
  h16*   XH   = (h16*)  alloc((size_t)M*64*2);
  float* RAW  = (float*)alloc((size_t)M*64*4);
  float* STATS= (float*)alloc(4096);
  h16*   LNXH = (h16*)  alloc((size_t)M*64*2);
  float* LR   = (float*)alloc((size_t)M*256*4);
  h16*   XLH  = (h16*)  alloc((size_t)M*128*2);
  float* DBC  = (float*)alloc((size_t)M*160*4);
  float* Yb   = (float*)alloc((size_t)M*128*4);
  h16*   ZH   = (h16*)  alloc((size_t)M*128*2);
  float* MAXB = (float*)alloc((size_t)M*64*4);
  float* ACC  = (float*)alloc((size_t)M*64*4);
  h16*   WTAP = (h16*)  alloc((size_t)2*27*4096*2);
  h16*   LRW  = (h16*)  alloc(256*64*2);
  h16*   WD   = (h16*)  alloc(160*128*2);
  h16*   OUTW = (h16*)  alloc(64*128*2);
  if (off > ws_size) return;   // workspace too small; bail safely

  dim3 b256(256);
  int gMC = (M*64 + 255)/256;      // 16384
  int gM  = (M + 255)/256;         // 256
  int gMD = (M*128 + 255)/256;     // 32768

  k_to_cl<<<gMC, b256, 0, stream>>>(x, XCL, XH);
  k_prep<<<1040, b256, 0, stream>>>(cr1_w, cr2_w,
                                    (const float*)d_in[7], (const float*)d_in[15],
                                    (const float*)d_in[10], (const float*)d_in[12],
                                    (const float*)d_in[13], (const float*)d_in[18],
                                    WTAP, LRW, WD, OUTW);
  // conv-res block 1
  k_conv<<<2048, b256, 0, stream>>>(XH, WTAP, cr1_b, RAW);
  k_stats<<<128, b256, 0, stream>>>(RAW, STATS);
  k_conv_fin<<<gMC, b256, 0, stream>>>(RAW, STATS, XCL, XH);
  // conv-res block 2
  k_conv<<<2048, b256, 0, stream>>>(XH, WTAP + (size_t)27*4096, cr2_b, RAW);
  k_stats<<<128, b256, 0, stream>>>(RAW, STATS);
  k_conv_fin<<<gMC, b256, 0, stream>>>(RAW, STATS, XCL, XH);
  // shared layernorm + left/right projections (axis independent)
  k_ln64<<<gM, b256, 0, stream>>>(XCL, ln_w, ln_b, LNXH);
  k_gemm<64,256><<<M/128, b256, 0, stream>>>(LNXH, LRW, LR);
  k_silu_gate<<<gMD, b256, 0, stream>>>(LR);

  const int strides[3] = {1024, 32, 1};   // H, W, D scan axes
  for (int a = 0; a < 3; a++){
    k_conv1d<<<gMD, b256, 0, stream>>>(LR, c1w, c1b, XLH, strides[a]);
    k_gemm<128,160><<<M/128, b256, 0, stream>>>(XLH, WD, DBC);
    k_scan<<<2048, dim3(128), 0, stream>>>(DBC, XLH, delta_b, A_log, Yb, a);
    k_pn<<<gM, b256, 0, stream>>>(Yb, LR, pn_w, pn_b, ZH);
    k_gemm<128,64><<<M/128, b256, 0, stream>>>(ZH, OUTW, MAXB);
    k_axpy<<<gMC, b256, 0, stream>>>(MAXB, axis_w, ACC, a, a == 0);
  }
  k_final<<<gMC, b256, 0, stream>>>(x, ACC, res_s, out);
}
